// CausalSelfAttention_17368847745133
// MI455X (gfx1250) — matrix operand order repaired and checked by reading, not compiled, not run on hardware
//
#include <hip/hip_runtime.h>
#include <hip/hip_bf16.h>

// ---------------------------------------------------------------------------
// CDNA5 (gfx1250) causal self-attention, bf16 WMMA + Tensor Data Mover.
//   x[B,T,C] --(GEMM1 WMMA, TDM-staged)--> q,k [B,H,T,D], v [B,H,D,T] bf16
//   flash-attention (QK^T, online softmax, P*V) with v_wmma_f32_16x16x32_bf16
//   y[B,T,C] bf16 --(GEMM2 WMMA, TDM-staged)--> out f32 [B,T,C]
// Weights are pre-transposed to [N][K] during f32->bf16 conversion so every
// LDS tile is a plain row-major 2D tile (TDM-describable, no LDS scatter).
// ---------------------------------------------------------------------------

typedef __bf16 bf16;
typedef __attribute__((ext_vector_type(16))) __bf16 v16bf;
typedef __attribute__((ext_vector_type(8)))  __bf16 v8bf;
typedef __attribute__((ext_vector_type(8)))  float  v8f;
typedef __attribute__((ext_vector_type(4)))  unsigned int tdm_g0_t;
typedef __attribute__((ext_vector_type(8)))  int          tdm_g1_t;
typedef __attribute__((ext_vector_type(4)))  int          tdm_g2_t;

#define N_EMBD 1024
#define N_HEAD 16
#define HEAD_D 64
#define BATCH  4
#define SEQ_T  2048
#define ROWS_M (BATCH * SEQ_T)   // 8192

#if defined(__has_builtin)
#  if __has_builtin(__builtin_amdgcn_tensor_load_to_lds)
#    define HAVE_TDM 1
#  endif
#endif
#ifndef HAVE_TDM
#  define HAVE_TDM 0
#endif

__device__ __forceinline__ bf16 f2bf(float f) {
  unsigned u = __builtin_bit_cast(unsigned, f);
  u += 0x7fffu + ((u >> 16) & 1u);              // round-to-nearest-even
  unsigned short h = (unsigned short)(u >> 16);
  return __builtin_bit_cast(bf16, h);
}

__device__ __forceinline__ unsigned lds_off(const void* p) {
  // LDS aperture keeps the byte offset in the low 32 bits of the generic VA.
  return (unsigned)(unsigned long long)p;
}

// ---------------------------------------------------------------------------
// TDM 2D tile load: rows x row_elems (bf16) from global (row stride
// stride_elems) into LDS at lds_byte, padding pad_amount dwords after each
// 2^(pad_interval+1) dwords so LDS rows land on a padded stride.
// Issued by one wave; completion via s_wait_tensorcnt + workgroup barrier.
// ---------------------------------------------------------------------------
__device__ __forceinline__ void tdm_load_2d(const bf16* gptr, unsigned lds_byte,
                                            unsigned rows, unsigned row_elems,
                                            unsigned stride_elems,
                                            unsigned pad_interval_code,
                                            unsigned pad_amount_code) {
#if HAVE_TDM
  unsigned long long ga = (unsigned long long)gptr;
  tdm_g0_t g0;
  g0[0] = 1u;                                          // count=1, user mode
  g0[1] = lds_byte;                                    // lds_addr
  g0[2] = (unsigned)(ga & 0xffffffffu);                // global_addr lo
  g0[3] = (unsigned)((ga >> 32) & 0x01ffffffu) | (2u << 30);  // addr hi | type=2
  unsigned d0 = (1u << 16)                             // data_size = 2 bytes
              | (1u << 20)                             // pad_enable
              | (pad_interval_code << 22)
              | (pad_amount_code << 25);
  unsigned tdim0 = stride_elems;                       // bound >= tile row
  unsigned tdim1 = 0x00100000u;                        // large: no dim1 OOB
  tdm_g1_t g1;
  g1[0] = (int)d0;
  g1[1] = (int)((tdim0 & 0xffffu) << 16);              // tensor_dim0[15:0]
  g1[2] = (int)((tdim0 >> 16) | ((tdim1 & 0xffffu) << 16));
  g1[3] = (int)((tdim1 >> 16) | (row_elems << 16));    // tile_dim0
  g1[4] = (int)(rows & 0xffffu);                       // tile_dim1 (tile_dim2=0)
  g1[5] = (int)stride_elems;                           // tensor_dim0_stride lo
  g1[6] = 0;
  g1[7] = 0;
  tdm_g2_t gz = {0, 0, 0, 0};
#if __clang_major__ >= 23
  tdm_g1_t gz8 = {0, 0, 0, 0, 0, 0, 0, 0};
  __builtin_amdgcn_tensor_load_to_lds(g0, g1, gz, gz, gz8, 0);
#else
  __builtin_amdgcn_tensor_load_to_lds(g0, g1, gz, gz, 0);
#endif
#else
  (void)gptr; (void)lds_byte; (void)rows; (void)row_elems;
  (void)stride_elems; (void)pad_interval_code; (void)pad_amount_code;
#endif
}

__device__ __forceinline__ void tdm_wait_all() {
#if HAVE_TDM
  __builtin_amdgcn_s_wait_tensorcnt((short)0);
#endif
}

// ---------------------------------------------------------------------------
// f32 -> bf16 elementwise convert
// ---------------------------------------------------------------------------
__global__ void f32_to_bf16_kernel(const float* __restrict__ src,
                                   bf16* __restrict__ dst, int n) {
  int i = blockIdx.x * blockDim.x + threadIdx.x;
  int stride = gridDim.x * blockDim.x;
  for (; i < n; i += stride) dst[i] = f2bf(src[i]);
}

// ---------------------------------------------------------------------------
// f32 [R][C] -> bf16 [C][R] tiled transpose (weights become [N][K])
// ---------------------------------------------------------------------------
__global__ __launch_bounds__(256) void transpose_f32_to_bf16(
    const float* __restrict__ src, bf16* __restrict__ dst, int R, int C) {
  __shared__ float tile[32][33];
  const int rb = blockIdx.y * 32, cb = blockIdx.x * 32;
  const int tx = threadIdx.x & 31, ty = threadIdx.x >> 5;
#pragma unroll
  for (int i = 0; i < 32; i += 8)
    tile[ty + i][tx] = src[(size_t)(rb + ty + i) * C + cb + tx];
  __syncthreads();
#pragma unroll
  for (int i = 0; i < 32; i += 8)
    dst[(size_t)(cb + ty + i) * R + rb + tx] = f2bf(tile[tx][ty + i]);
}

// ---------------------------------------------------------------------------
// Tiled bf16 WMMA GEMM: C[M,N] = A[M,K] * Bt[N,K]^T + bias[N]
//   block tile 64(M) x 128(N), K-step 32; 8 waves, wave computes 32x32.
//   Double-buffered LDS; tiles staged by the Tensor Data Mover.
//   mode 0: scatter bf16 into q/k [B,H,T,D] and v [B,H,D,T]
//   mode 1: f32 result (final projection) to outF [M,N]
// ---------------------------------------------------------------------------
#define GBM 64
#define GBN 128
#define GBK 32
#define GLD 40   // padded LDS row stride (halfs): 80 B (TDM pad 16B / 64B row)

__global__ __launch_bounds__(256) void gemm_bf16_wmma(
    const bf16* __restrict__ A, const bf16* __restrict__ Bt,
    const float* __restrict__ bias, int M, int N, int K, int mode,
    bf16* __restrict__ qO, bf16* __restrict__ kO, bf16* __restrict__ vO,
    float* __restrict__ outF)
{
  __shared__ alignas(16) bf16 As[2][GBM][GLD];
  __shared__ alignas(16) bf16 Bs[2][GBN][GLD];   // Bs[n][k] (Bt row-major)

  const int tid  = threadIdx.x;
  const int wave = tid >> 5;
  const int lane = tid & 31;
  const int l15  = lane & 15;
  const int hg   = lane >> 4;

  const int mBase = blockIdx.x * GBM;
  const int nBase = blockIdx.y * GBN;
  const int m0 = (wave & 1) * 32;
  const int n0 = (wave >> 1) * 32;

  v8f acc[2][2];
#pragma unroll
  for (int mt = 0; mt < 2; ++mt)
#pragma unroll
    for (int nt = 0; nt < 2; ++nt)
#pragma unroll
      for (int r = 0; r < 8; ++r) acc[mt][nt][r] = 0.f;

  auto stage = [&](int k0, int buf) {
#if HAVE_TDM
    if (wave == 0) {
      tdm_load_2d(A  + (size_t)mBase * K + k0, lds_off(&As[buf][0][0]),
                  GBM, GBK, (unsigned)K, 3u, 3u);      // 64B rows, pad 16B
      tdm_load_2d(Bt + (size_t)nBase * K + k0, lds_off(&Bs[buf][0][0]),
                  GBN, GBK, (unsigned)K, 3u, 3u);
    }
#else
    {
      int row = tid >> 2, c = (tid & 3) * 8;
      *(v8bf*)(&As[buf][row][c]) =
          *(const v8bf*)(A + (size_t)(mBase + row) * K + k0 + c);
    }
#pragma unroll
    for (int cc = tid; cc < 512; cc += 256) {
      int row = cc >> 2, c = (cc & 3) * 8;
      *(v8bf*)(&Bs[buf][row][c]) =
          *(const v8bf*)(Bt + (size_t)(nBase + row) * K + k0 + c);
    }
    if (k0 + GBK < K)
      __builtin_prefetch(A + (size_t)mBase * K + k0 + GBK, 0, 1);
#endif
  };

  stage(0, 0);
  tdm_wait_all();
  __syncthreads();

  int cur = 0;
  for (int k0 = 0; k0 < K; k0 += GBK) {
    if (k0 + GBK < K) stage(k0 + GBK, cur ^ 1);   // DMA next tile during compute

    v16bf af[2], bfm[2];
#pragma unroll
    for (int mt = 0; mt < 2; ++mt) {
      int row = m0 + mt * 16 + l15;
      int ka  = hg * 8;                       // A layout: lane<16 K{0-7,16-23}
      v8bf lo = *(const v8bf*)(&As[cur][row][ka]);
      v8bf hi = *(const v8bf*)(&As[cur][row][ka + 16]);
#pragma unroll
      for (int i = 0; i < 8; ++i) { af[mt][i] = lo[i]; af[mt][i + 8] = hi[i]; }
    }
#pragma unroll
    for (int nt = 0; nt < 2; ++nt) {
      int col = n0 + nt * 16 + l15;
      int kb  = hg * 8;                       // B layout: lane<16 K{0-7,16-23}
      v8bf lo = *(const v8bf*)(&Bs[cur][col][kb]);
      v8bf hi = *(const v8bf*)(&Bs[cur][col][kb + 16]);
#pragma unroll
      for (int i = 0; i < 8; ++i) { bfm[nt][i] = lo[i]; bfm[nt][i + 8] = hi[i]; }
    }
#pragma unroll
    for (int mt = 0; mt < 2; ++mt)
#pragma unroll
      for (int nt = 0; nt < 2; ++nt)
        acc[mt][nt] = __builtin_amdgcn_wmma_f32_16x16x32_bf16(
            false, af[mt], false, bfm[nt], (short)0, acc[mt][nt], false, false);

    tdm_wait_all();        // wave0: DMA of next tile complete
    __syncthreads();       // everyone: done reading cur, next tile visible
    cur ^= 1;
  }

  // epilogue: C/D layout -> M = r + 8*hg, N = l15 -----------------------------
#pragma unroll
  for (int mt = 0; mt < 2; ++mt)
#pragma unroll
    for (int nt = 0; nt < 2; ++nt)
#pragma unroll
      for (int r = 0; r < 8; ++r) {
        int gm = mBase + m0 + mt * 16 + hg * 8 + r;
        int gn = nBase + n0 + nt * 16 + l15;
        float v = acc[mt][nt][r] + bias[gn];
        if (mode == 0) {
          int b = gm >> 11, t = gm & (SEQ_T - 1);
          int which = gn >> 10, rem = gn & 1023;
          int h = rem >> 6, d = rem & 63;
          bf16 bv = f2bf(v);
          if (which == 0) {
            qO[((((size_t)b * N_HEAD + h) * SEQ_T + t) << 6) + d] = bv;
          } else if (which == 1) {
            kO[((((size_t)b * N_HEAD + h) * SEQ_T + t) << 6) + d] = bv;
          } else {        // V stored pre-transposed: [B,H,D,T]
            vO[(((size_t)b * N_HEAD + h) * HEAD_D + d) * SEQ_T + t] = bv;
          }
        } else {
          outF[(size_t)gm * N + gn] = v;
        }
      }
}

// ---------------------------------------------------------------------------
// Flash attention: one workgroup per (b*h, 128-row q block).
// Each wave owns 16 full q rows -> softmax reductions stay in a half-wave.
// K [key][d] and V^T [d][key] tiles are plain 2D tiles -> TDM staged.
// ---------------------------------------------------------------------------
#define QB 128
#define KB 64
#define ALD 72   // padded LDS stride (halfs): 144 B (TDM pad 16B / 128B row)

__global__ __launch_bounds__(256) void flash_attn_wmma(
    const bf16* __restrict__ Qg, const bf16* __restrict__ Kg,
    const bf16* __restrict__ Vtg, const int* __restrict__ lens,
    bf16* __restrict__ Y)
{
  __shared__ alignas(16) bf16 Qs[QB][ALD];
  __shared__ alignas(16) bf16 Ks[KB][ALD];          // [key][d]
  __shared__ alignas(16) bf16 Vt[HEAD_D][ALD];      // [d][key]
  __shared__ alignas(16) bf16 Ps[8][16][ALD];       // per-wave P re-layout

  const int bh = blockIdx.x;
  const int qb = blockIdx.y;
  const int b  = bh >> 4;
  const int h  = bh & 15;
  const int lb = lens[b];

  const int tid  = threadIdx.x;
  const int wave = tid >> 5;
  const int lane = tid & 31;
  const int l15  = lane & 15;
  const int hg   = lane >> 4;

  const size_t base   = (size_t)bh * SEQ_T * HEAD_D;   // q/k [T][D]
  const size_t baseVt = (size_t)bh * HEAD_D * SEQ_T;   // v  [D][T]

  // stage q block once ------------------------------------------------------
#if HAVE_TDM
  if (wave == 0)
    tdm_load_2d(Qg + base + (size_t)qb * QB * HEAD_D, lds_off(&Qs[0][0]),
                QB, HEAD_D, HEAD_D, 4u, 3u);           // 128B rows, pad 16B
  tdm_wait_all();
#else
  for (int c = tid; c < (QB * HEAD_D / 8); c += 256) {
    int row = c >> 3, off = (c & 7) * 8;
    *(v8bf*)(&Qs[row][off]) =
        *(const v8bf*)(Qg + base + (size_t)(qb * QB + row) * HEAD_D + off);
  }
#endif

  float m_r[8], l_r[8];
  v8f o[4];
#pragma unroll
  for (int r = 0; r < 8; ++r) { m_r[r] = -3.0e38f; l_r[r] = 0.f; }
#pragma unroll
  for (int nt = 0; nt < 4; ++nt)
#pragma unroll
    for (int r = 0; r < 8; ++r) o[nt][r] = 0.f;

  const float scale = 0.125f;             // 1/sqrt(64)
  const int   nkb   = 2 * qb + 2;
  const int   qrow0 = qb * QB + wave * 16;

  for (int kb = 0; kb < nkb; ++kb) {
    __syncthreads();                      // prior P*V done before LDS reuse
#if HAVE_TDM
    if (wave == 0) {
      tdm_load_2d(Kg + base + (size_t)kb * KB * HEAD_D, lds_off(&Ks[0][0]),
                  KB, HEAD_D, HEAD_D, 4u, 3u);
      tdm_load_2d(Vtg + baseVt + (size_t)kb * KB, lds_off(&Vt[0][0]),
                  HEAD_D, KB, SEQ_T, 4u, 3u);
    }
    tdm_wait_all();
#else
    for (int c = tid; c < (KB * HEAD_D / 8); c += 256) {
      int row = c >> 3, off = (c & 7) * 8;
      *(v8bf*)(&Ks[row][off]) =
          *(const v8bf*)(Kg + base + (size_t)(kb * KB + row) * HEAD_D + off);
    }
    for (int c = tid; c < (KB * HEAD_D / 8); c += 256) {
      int row = c >> 3, off = (c & 7) * 8;   // row = d, cols = keys
      *(v8bf*)(&Vt[row][off]) =
          *(const v8bf*)(Vtg + baseVt + (size_t)row * SEQ_T + kb * KB + off);
    }
#endif
    __syncthreads();

    // S = Q * K^T : 16 q rows x 64 keys, K-dim 64 (2 wmma steps) ------------
    v8f st[4];
#pragma unroll
    for (int nt = 0; nt < 4; ++nt)
#pragma unroll
      for (int r = 0; r < 8; ++r) st[nt][r] = 0.f;

#pragma unroll
    for (int ks = 0; ks < 2; ++ks) {
      v16bf aq;
      {
        int row = wave * 16 + l15;
        int ka  = ks * 32 + hg * 8;
        v8bf lo = *(const v8bf*)(&Qs[row][ka]);
        v8bf hi = *(const v8bf*)(&Qs[row][ka + 16]);
#pragma unroll
        for (int i = 0; i < 8; ++i) { aq[i] = lo[i]; aq[i + 8] = hi[i]; }
      }
#pragma unroll
      for (int nt = 0; nt < 4; ++nt) {
        int key = nt * 16 + l15;
        int kk  = ks * 32 + hg * 8;
        v8bf lo = *(const v8bf*)(&Ks[key][kk]);
        v8bf hi = *(const v8bf*)(&Ks[key][kk + 16]);
        v16bf bk;
#pragma unroll
        for (int i = 0; i < 8; ++i) { bk[i] = lo[i]; bk[i + 8] = hi[i]; }
        st[nt] = __builtin_amdgcn_wmma_f32_16x16x32_bf16(
            false, aq, false, bk, (short)0, st[nt], false, false);
      }
    }

    // additive mask + online softmax ----------------------------------------
#pragma unroll
    for (int r = 0; r < 8; ++r) {
      int qg = qrow0 + hg * 8 + r;
      float sv[4];
      float rmax = -3.0e38f;
#pragma unroll
      for (int nt = 0; nt < 4; ++nt) {
        int kg = kb * KB + nt * 16 + l15;
        bool keep = (kg <= qg) && (kg < lb) && (qg < lb);
        float s = st[nt][r] * scale + (keep ? 0.f : -1.0e8f);
        sv[nt] = s;
        rmax = fmaxf(rmax, s);
      }
#pragma unroll
      for (int mk = 1; mk <= 8; mk <<= 1)
        rmax = fmaxf(rmax, __shfl_xor(rmax, mk, 32));
      float mn = fmaxf(m_r[r], rmax);
      float alpha = __expf(m_r[r] - mn);
      float rsum = 0.f;
#pragma unroll
      for (int nt = 0; nt < 4; ++nt) {
        float p = __expf(sv[nt] - mn);
        st[nt][r] = p;
        rsum += p;
      }
#pragma unroll
      for (int mk = 1; mk <= 8; mk <<= 1) rsum += __shfl_xor(rsum, mk, 32);
      l_r[r] = l_r[r] * alpha + rsum;
      m_r[r] = mn;
#pragma unroll
      for (int nt = 0; nt < 4; ++nt) o[nt][r] *= alpha;
    }

    // P: accumulator layout -> A-fragment layout via per-wave LDS -----------
#pragma unroll
    for (int nt = 0; nt < 4; ++nt)
#pragma unroll
      for (int r = 0; r < 8; ++r)
        Ps[wave][hg * 8 + r][nt * 16 + l15] = f2bf(st[nt][r]);
    __syncthreads();

    // O += P * V : K-dim 64 keys (2 wmma steps), N = 64 d (4 tiles) ---------
#pragma unroll
    for (int ks = 0; ks < 2; ++ks) {
      v16bf ap;
      {
        int ka = ks * 32 + hg * 8;
        v8bf lo = *(const v8bf*)(&Ps[wave][l15][ka]);
        v8bf hi = *(const v8bf*)(&Ps[wave][l15][ka + 16]);
#pragma unroll
        for (int i = 0; i < 8; ++i) { ap[i] = lo[i]; ap[i + 8] = hi[i]; }
      }
#pragma unroll
      for (int nt = 0; nt < 4; ++nt) {
        int d  = nt * 16 + l15;
        int kk = ks * 32 + hg * 8;
        v8bf lo = *(const v8bf*)(&Vt[d][kk]);
        v8bf hi = *(const v8bf*)(&Vt[d][kk + 16]);
        v16bf bv;
#pragma unroll
        for (int i = 0; i < 8; ++i) { bv[i] = lo[i]; bv[i + 8] = hi[i]; }
        o[nt] = __builtin_amdgcn_wmma_f32_16x16x32_bf16(
            false, ap, false, bv, (short)0, o[nt], false, false);
      }
    }
  }

  // normalize + store y[B,T,C] bf16 ------------------------------------------
#pragma unroll
  for (int nt = 0; nt < 4; ++nt)
#pragma unroll
    for (int r = 0; r < 8; ++r) {
      int qg = qrow0 + hg * 8 + r;
      int d  = nt * 16 + l15;
      float v = o[nt][r] / l_r[r];
      Y[((size_t)b * SEQ_T + qg) * N_EMBD + h * HEAD_D + d] = f2bf(v);
    }
}

// ---------------------------------------------------------------------------
// launch
// ---------------------------------------------------------------------------
extern "C" void kernel_launch(void* const* d_in, const int* in_sizes, int n_in,
                              void* d_out, int out_size, void* d_ws, size_t ws_size,
                              hipStream_t stream) {
  const float* x      = (const float*)d_in[0];
  const int*   lens   = (const int*)  d_in[1];
  const float* W_attn = (const float*)d_in[2];
  const float* b_attn = (const float*)d_in[3];
  const float* W_proj = (const float*)d_in[4];
  const float* b_proj = (const float*)d_in[5];
  float* out = (float*)d_out;

  char* w = (char*)d_ws;
  size_t off = 0;
  auto alloc = [&](size_t bytes) -> void* {
    off = (off + 255) & ~(size_t)255;
    void* p = w + off;
    off += bytes;
    return p;
  };
  const size_t nX  = (size_t)ROWS_M * N_EMBD;
  const size_t nWa = (size_t)N_EMBD * 3 * N_EMBD;
  const size_t nWp = (size_t)N_EMBD * N_EMBD;
  const size_t nHD = (size_t)BATCH * N_HEAD * SEQ_T * HEAD_D;

  bf16* xbf  = (bf16*)alloc(nX * 2);
  bf16* wabf = (bf16*)alloc(nWa * 2);   // transposed: [3C][C]
  bf16* wpbf = (bf16*)alloc(nWp * 2);   // transposed: [C][C]
  bf16* qbf  = (bf16*)alloc(nHD * 2);   // [B,H,T,D]
  bf16* kbf  = (bf16*)alloc(nHD * 2);   // [B,H,T,D]
  bf16* vbf  = (bf16*)alloc(nHD * 2);   // [B,H,D,T]  (pre-transposed)
  bf16* ybf  = (bf16*)alloc(nX * 2);    // [B,T,C]
  (void)ws_size; (void)n_in; (void)in_sizes; (void)out_size;

  // 1) conversions: x elementwise; weights transposed to [N][K]
  f32_to_bf16_kernel<<<2048, 256, 0, stream>>>(x, xbf, (int)nX);
  transpose_f32_to_bf16<<<dim3(3 * N_EMBD / 32, N_EMBD / 32), 256, 0, stream>>>(
      W_attn, wabf, N_EMBD, 3 * N_EMBD);
  transpose_f32_to_bf16<<<dim3(N_EMBD / 32, N_EMBD / 32), 256, 0, stream>>>(
      W_proj, wpbf, N_EMBD, N_EMBD);

  // 2) QKV projection: [8192,1024] x [1024,3072] -> q/k/v (v transposed)
  gemm_bf16_wmma<<<dim3(ROWS_M / GBM, (3 * N_EMBD) / GBN), 256, 0, stream>>>(
      xbf, wabf, b_attn, ROWS_M, 3 * N_EMBD, N_EMBD, 0,
      qbf, kbf, vbf, nullptr);

  // 3) flash attention: grid (B*H, T/128)
  flash_attn_wmma<<<dim3(BATCH * N_HEAD, SEQ_T / QB), 256, 0, stream>>>(
      qbf, kbf, vbf, lens, ybf);

  // 4) output projection: [8192,1024] x [1024,1024] + bias -> f32 out
  gemm_bf16_wmma<<<dim3(ROWS_M / GBM, N_EMBD / GBN), 256, 0, stream>>>(
      ybf, wpbf, b_proj, ROWS_M, N_EMBD, N_EMBD, 1,
      nullptr, nullptr, nullptr, out);
}
